// Net_18872086298646
// MI455X (gfx1250) — matrix operand order repaired and checked by reading, not compiled, not run on hardware
//
#include <hip/hip_runtime.h>
#include <hip/hip_bf16.h>
#include <stdint.h>
#include <stddef.h>

#define NNODE  20000
#define NEDGE  320000
#define EPLUS  (NEDGE + NNODE)   // edges + self loops
#define NGRAPH 256
#define HEADS  4
#define SLOPE  0.2f

typedef __bf16 bf16_t;
typedef __attribute__((ext_vector_type(16))) __bf16 v16bf;
typedef __attribute__((ext_vector_type(8)))  __bf16 v8bf;
typedef __attribute__((ext_vector_type(8)))  float  v8f;

// ---------- helpers ----------
__device__ __forceinline__ bf16_t f2bf(float f) {
  unsigned u = __float_as_uint(f);
  u = (u + 0x7FFFu + ((u >> 16) & 1u)) >> 16;   // round-to-nearest-even
  unsigned short s = (unsigned short)u;
  bf16_t b; __builtin_memcpy(&b, &s, 2); return b;
}
// order-preserving float<->uint for atomicMax
__device__ __forceinline__ unsigned fenc(float f) {
  unsigned u = __float_as_uint(f);
  return (u & 0x80000000u) ? ~u : (u | 0x80000000u);
}
__device__ __forceinline__ float fdec(unsigned u) {
  u = (u & 0x80000000u) ? (u & 0x7FFFFFFFu) : ~u;
  return __uint_as_float(u);
}
__device__ __forceinline__ void edge_sd(const int* __restrict__ eidx, int e, int& s, int& d) {
  if (e < NEDGE) { s = eidx[e]; d = eidx[NEDGE + e]; }
  else           { s = d = e - NEDGE; }          // self loop
}
__device__ __forceinline__ float lrelu(float l) { return l > 0.f ? l : SLOPE * l; }

// ---------- generic kernels ----------
__global__ void zero_k(float* __restrict__ p, int n) {
  int i = blockIdx.x * blockDim.x + threadIdx.x;
  if (i < n) p[i] = 0.0f;
}
__global__ void cvt_bf16_k(const float* __restrict__ in, bf16_t* __restrict__ out, int n) {
  int i = blockIdx.x * blockDim.x + threadIdx.x;
  if (i < n) out[i] = f2bf(in[i]);
}

// ---------- WMMA GEMM ----------
// C[Mrows x Ncols] = A[Mrows x K](bf16,row-major) * B[K x Ncols](bf16,row-major) (+bias)
// block = 128 threads = 4 waves = 2 row-tiles x 2 col-groups.
// Each wave: 16 rows x 64 cols = 4 accumulator tiles sharing one A fragment per K-step.
// Software-pipelined: next K-step's fragments are loaded before this step's WMMAs.
// grid.x = Mrows/32, grid.y = Ncols/128.
__global__ __launch_bounds__(128, 1)
void wmma_gemm_k(const bf16_t* __restrict__ A, const bf16_t* __restrict__ B,
                 const float* __restrict__ bias, float* __restrict__ C,
                 int K, int Ncols)
{
  const int wave = threadIdx.x >> 5;
  const int wrow = wave >> 1;           // 0..1 row tile
  const int wcol = wave & 1;            // 0..1 col group
  const int lane = threadIdx.x & 31;
  const int half = lane >> 4;           // 0|1 lane group
  const int l16  = lane & 15;
  const int row0 = blockIdx.x * 32 + wrow * 16;
  const int n0   = blockIdx.y * 128 + wcol * 64;

  const bf16_t* arow = A + (size_t)(row0 + l16) * K + half * 8;  // + k0 per step
  const bf16_t* bcol = B + n0 + l16;                             // + k*Ncols + t*16 per element

  auto load_a = [&](int k0) -> v16bf {
    const bf16_t* ap = arow + k0;
    v8bf a0 = *(const v8bf*)ap;
    v8bf a1 = *(const v8bf*)(ap + 16);
    v16bf af;
#pragma unroll
    for (int i = 0; i < 8; ++i) { af[i] = a0[i]; af[i + 8] = a1[i]; }
    return af;
  };
  auto load_b = [&](int k0, int t) -> v16bf {
    const bf16_t* bp = bcol + t * 16;
    v16bf bf;
#pragma unroll
    for (int i = 0; i < 8; ++i) {
      bf[i]     = bp[(size_t)(k0 + half * 8 + i) * Ncols];
      bf[i + 8] = bp[(size_t)(k0 + 16 + half * 8 + i) * Ncols];
    }
    return bf;
  };

  v8f acc0 = {}, acc1 = {}, acc2 = {}, acc3 = {};

  // prologue: fragments for k0 = 0
  v16bf af = load_a(0);
  v16bf b0 = load_b(0, 0);
  v16bf b1 = load_b(0, 1);
  v16bf b2 = load_b(0, 2);
  v16bf b3 = load_b(0, 3);

  for (int k0 = 0; k0 < K; k0 += 32) {
    // prefetch next K-step (address clamped in-bounds on last iteration; data unused)
    const int kn = (k0 + 32 < K) ? (k0 + 32) : 0;
    v16bf afn = load_a(kn);
    v16bf b0n = load_b(kn, 0);
    v16bf b1n = load_b(kn, 1);
    v16bf b2n = load_b(kn, 2);
    v16bf b3n = load_b(kn, 3);

    acc0 = __builtin_amdgcn_wmma_f32_16x16x32_bf16(false, af, false, b0, (short)0, acc0, false, false);
    acc1 = __builtin_amdgcn_wmma_f32_16x16x32_bf16(false, af, false, b1, (short)0, acc1, false, false);
    acc2 = __builtin_amdgcn_wmma_f32_16x16x32_bf16(false, af, false, b2, (short)0, acc2, false, false);
    acc3 = __builtin_amdgcn_wmma_f32_16x16x32_bf16(false, af, false, b3, (short)0, acc3, false, false);

    af = afn; b0 = b0n; b1 = b1n; b2 = b2n; b3 = b3n;
  }

  // epilogue: C layout -> element r maps to (M = r + 8*half, N = lane%16)
  v8f accs[4] = {acc0, acc1, acc2, acc3};
#pragma unroll
  for (int t = 0; t < 4; ++t) {
    const int n  = n0 + t * 16 + l16;
    const float bv = bias ? bias[n] : 0.0f;
#pragma unroll
    for (int r = 0; r < 8; ++r) {
      const int m = row0 + half * 8 + r;
      C[(size_t)m * Ncols + n] = accs[t][r] + bv;
    }
  }
}

// ---------- GAT-specific kernels ----------
__global__ void embed_k(const int* __restrict__ xi, const float* __restrict__ emb,
                        bf16_t* __restrict__ xb) {
  int i = blockIdx.x * blockDim.x + threadIdx.x;            // N*128
  if (i >= NNODE * 128) return;
  int node = i >> 7, c = i & 127;
  xb[i] = f2bf(emb[xi[node] * 128 + c]);
}

// per-(node,head) attention projections: a = <xh[n,h,:], att[h,:]>
__global__ void attn_k(const float* __restrict__ xh, const float* __restrict__ atts,
                       const float* __restrict__ attd, float* __restrict__ a_s,
                       float* __restrict__ a_d, int C) {
  int i = blockIdx.x * blockDim.x + threadIdx.x;            // N*HEADS
  if (i >= NNODE * HEADS) return;
  int n = i >> 2, h = i & 3;
  const float4* xp = (const float4*)(xh + ((size_t)n * HEADS + h) * C);
  const float4* sp = (const float4*)(atts + h * C);
  const float4* dp = (const float4*)(attd + h * C);
  float s = 0.f, d = 0.f;
  for (int c = 0; c < (C >> 2); ++c) {
    float4 v = xp[c], a = sp[c], b = dp[c];
    s += v.x * a.x + v.y * a.y + v.z * a.z + v.w * a.w;
    d += v.x * b.x + v.y * b.y + v.z * b.z + v.w * b.w;
  }
  a_s[i] = s; a_d[i] = d;
}

__global__ void edge_max_k(const int* __restrict__ eidx, const float* __restrict__ a_s,
                           const float* __restrict__ a_d, unsigned* __restrict__ menc) {
  int e = blockIdx.x * blockDim.x + threadIdx.x;
  if (e >= EPLUS) return;
  if (e + 8192 < NEDGE) {                       // stream-ahead hint for the index arrays
    __builtin_prefetch(&eidx[e + 8192], 0, 1);
    __builtin_prefetch(&eidx[NEDGE + e + 8192], 0, 1);
  }
  int s, d; edge_sd(eidx, e, s, d);
#pragma unroll
  for (int h = 0; h < HEADS; ++h) {
    float l = lrelu(a_s[s * HEADS + h] + a_d[d * HEADS + h]);
    atomicMax(&menc[d * HEADS + h], fenc(l));
  }
}

__global__ void edge_sum_k(const int* __restrict__ eidx, const float* __restrict__ a_s,
                           const float* __restrict__ a_d, const unsigned* __restrict__ menc,
                           float* __restrict__ z) {
  int e = blockIdx.x * blockDim.x + threadIdx.x;
  if (e >= EPLUS) return;
  int s, d; edge_sd(eidx, e, s, d);
#pragma unroll
  for (int h = 0; h < HEADS; ++h) {
    float l = lrelu(a_s[s * HEADS + h] + a_d[d * HEADS + h]);
    atomicAdd(&z[d * HEADS + h], __expf(l - fdec(menc[d * HEADS + h])));
  }
}

// message scatter: gat[dst,h,c] += alpha(e,h) * xh[src,h,c].
// C/4 threads per edge; each thread moves a float4 per head.
__global__ void edge_msg_k(const int* __restrict__ eidx, const float* __restrict__ a_s,
                           const float* __restrict__ a_d, const unsigned* __restrict__ menc,
                           const float* __restrict__ z, const float* __restrict__ xh,
                           float* __restrict__ gat, int C) {
  int tpe = C >> 2;                        // threads per edge (16 for C=64, 32 for C=128)
  int epb = 256 / tpe;                     // edges per block
  int le  = threadIdx.x / tpe;
  int c0  = (threadIdx.x - le * tpe) << 2; // channel base (multiple of 4)
  int e   = blockIdx.x * epb + le;
  if (e >= EPLUS) return;
  int s, d; edge_sd(eidx, e, s, d);
#pragma unroll
  for (int h = 0; h < HEADS; ++h) {
    float l = lrelu(a_s[s * HEADS + h] + a_d[d * HEADS + h]);
    float alpha = __expf(l - fdec(menc[d * HEADS + h])) / (z[d * HEADS + h] + 1e-16f);
    float4 v = *(const float4*)&xh[((size_t)s * HEADS + h) * C + c0];
    float* gp = &gat[((size_t)d * HEADS + h) * C + c0];
    atomicAdd(gp + 0, alpha * v.x);
    atomicAdd(gp + 1, alpha * v.y);
    atomicAdd(gp + 2, alpha * v.z);
    atomicAdd(gp + 3, alpha * v.w);
  }
}

// layers 1-2: x = relu(gat + b + lin); emit bf16 for the next GEMM (4 elems/thread)
__global__ void finish12_k(const float* __restrict__ gat, const float* __restrict__ b,
                           const float* __restrict__ lin, bf16_t* __restrict__ xb) {
  int i4 = blockIdx.x * blockDim.x + threadIdx.x;           // N*256/4
  if (i4 >= NNODE * 64) return;
  int i = i4 << 2;
  int j = i & 255;
  float4 g = *(const float4*)&gat[i];
  float4 l = *(const float4*)&lin[i];
  float4 bb = *(const float4*)&b[j];
  float v0 = fmaxf(g.x + bb.x + l.x, 0.f);
  float v1 = fmaxf(g.y + bb.y + l.y, 0.f);
  float v2 = fmaxf(g.z + bb.z + l.z, 0.f);
  float v3 = fmaxf(g.w + bb.w + l.w, 0.f);
  bf16_t* o = xb + i;
  o[0] = f2bf(v0); o[1] = f2bf(v1); o[2] = f2bf(v2); o[3] = f2bf(v3);
}

// layer 3: xo = mean_h(gat) + b3 + lin3 (no relu)
__global__ void finish3_k(const float* __restrict__ gat, const float* __restrict__ b3,
                          const float* __restrict__ lin3, float* __restrict__ xo) {
  int i = blockIdx.x * blockDim.x + threadIdx.x;            // N*128
  if (i >= NNODE * 128) return;
  int n = i >> 7, j = i & 127;
  const float* g = gat + (size_t)n * 512;
  float v = 0.25f * (g[j] + g[128 + j] + g[256 + j] + g[384 + j]);
  xo[i] = v + b3[j] + lin3[i];
}

__global__ void pool_acc_k(const float* __restrict__ xo, const int* __restrict__ batch,
                           float* __restrict__ pooled, float* __restrict__ cnt) {
  int i = blockIdx.x * blockDim.x + threadIdx.x;            // N*128
  if (i >= NNODE * 128) return;
  int n = i >> 7, j = i & 127;
  int g = batch[n];
  atomicAdd(&pooled[g * 128 + j], xo[i]);
  if (j == 0) atomicAdd(&cnt[g], 1.0f);
}
__global__ void pool_div_k(const float* __restrict__ pooled, const float* __restrict__ cnt,
                           float* __restrict__ out) {
  int i = blockIdx.x * blockDim.x + threadIdx.x;            // G*128
  if (i >= NGRAPH * 128) return;
  out[i] = pooled[i] / fmaxf(cnt[i >> 7], 1.0f);
}

// ---------- launcher ----------
static inline int cdiv(int a, int b) { return (a + b - 1) / b; }

extern "C" void kernel_launch(void* const* d_in, const int* in_sizes, int n_in,
                              void* d_out, int out_size, void* d_ws, size_t ws_size,
                              hipStream_t stream) {
  (void)in_sizes; (void)n_in; (void)out_size; (void)ws_size;

  const int*   x_l     = (const int*)  d_in[0];
  const int*   edge_l  = (const int*)  d_in[1];
  const int*   batch_l = (const int*)  d_in[2];
  const int*   x_r     = (const int*)  d_in[3];
  const int*   edge_r  = (const int*)  d_in[4];
  const int*   batch_r = (const int*)  d_in[5];
  const float* emb     = (const float*)d_in[6];
  const float* W1  = (const float*)d_in[7],  *as1 = (const float*)d_in[8],
             * ad1 = (const float*)d_in[9],  *b1  = (const float*)d_in[10],
             * lw1 = (const float*)d_in[11], *lb1 = (const float*)d_in[12];
  const float* W2  = (const float*)d_in[13], *as2 = (const float*)d_in[14],
             * ad2 = (const float*)d_in[15], *b2  = (const float*)d_in[16],
             * lw2 = (const float*)d_in[17], *lb2 = (const float*)d_in[18];
  const float* W3  = (const float*)d_in[19], *as3 = (const float*)d_in[20],
             * ad3 = (const float*)d_in[21], *b3  = (const float*)d_in[22],
             * lw3 = (const float*)d_in[23], *lb3 = (const float*)d_in[24];
  float* out = (float*)d_out;

  // ---- workspace bump allocator ----
  char* ws = (char*)d_ws;
  size_t off = 0;
  auto alloc = [&](size_t bytes) -> void* {
    void* p = ws + off;
    off += (bytes + 255) & ~(size_t)255;
    return p;
  };
  bf16_t* Wb1 = (bf16_t*)alloc(128 * 256 * 2);
  bf16_t* Lb1 = (bf16_t*)alloc(128 * 256 * 2);
  bf16_t* Wb2 = (bf16_t*)alloc(256 * 256 * 2);
  bf16_t* Lb2 = (bf16_t*)alloc(256 * 256 * 2);
  bf16_t* Wb3 = (bf16_t*)alloc(256 * 512 * 2);
  bf16_t* Lb3 = (bf16_t*)alloc(256 * 128 * 2);
  bf16_t* xb   = (bf16_t*)alloc((size_t)NNODE * 256 * 2);
  float*  xh   = (float*) alloc((size_t)NNODE * 512 * 4);
  float*  lin  = (float*) alloc((size_t)NNODE * 256 * 4);
  float*  a_s  = (float*) alloc((size_t)NNODE * HEADS * 4);
  float*  a_d  = (float*) alloc((size_t)NNODE * HEADS * 4);
  unsigned* menc = (unsigned*)alloc((size_t)NNODE * HEADS * 4);
  float*  zbuf = (float*) alloc((size_t)NNODE * HEADS * 4);
  float*  gat  = (float*) alloc((size_t)NNODE * 512 * 4);
  float*  xo   = (float*) alloc((size_t)NNODE * 128 * 4);
  float*  pooled = (float*)alloc((size_t)NGRAPH * 128 * 4);
  float*  cnt    = (float*)alloc((size_t)NGRAPH * 4);

  auto zero = [&](float* p, int n) {
    zero_k<<<cdiv(n, 256), 256, 0, stream>>>(p, n);
  };
  auto cvt = [&](const float* p, bf16_t* o, int n) {
    cvt_bf16_k<<<cdiv(n, 256), 256, 0, stream>>>(p, o, n);
  };
  auto gemm = [&](const bf16_t* A, const bf16_t* B, const float* bias, float* C,
                  int K, int Ncols) {
    dim3 grid(NNODE / 32, Ncols / 128);   // 625 x {1,2,4}
    wmma_gemm_k<<<grid, 128, 0, stream>>>(A, B, bias, C, K, Ncols);
  };

  // weights -> bf16 (shared by both sides)
  cvt(W1,  Wb1, 128 * 256);  cvt(lw1, Lb1, 128 * 256);
  cvt(W2,  Wb2, 256 * 256);  cvt(lw2, Lb2, 256 * 256);
  cvt(W3,  Wb3, 256 * 512);  cvt(lw3, Lb3, 256 * 128);

  const int egrid = cdiv(EPLUS, 256);

  auto run_side = [&](const int* xi, const int* eidx, const int* batch, float* out_side) {
    // embedding -> bf16 node features [N,128]
    embed_k<<<cdiv(NNODE * 128, 256), 256, 0, stream>>>(xi, emb, xb);

    // ---- layer 1: K=128, H*C=256, C=64 ----
    gemm(xb, Wb1, nullptr, xh, 128, 256);
    gemm(xb, Lb1, lb1,    lin, 128, 256);
    attn_k<<<cdiv(NNODE * HEADS, 256), 256, 0, stream>>>(xh, as1, ad1, a_s, a_d, 64);
    zero((float*)menc, NNODE * HEADS); zero(zbuf, NNODE * HEADS); zero(gat, NNODE * 256);
    edge_max_k<<<egrid, 256, 0, stream>>>(eidx, a_s, a_d, menc);
    edge_sum_k<<<egrid, 256, 0, stream>>>(eidx, a_s, a_d, menc, zbuf);
    edge_msg_k<<<cdiv(EPLUS, 16), 256, 0, stream>>>(eidx, a_s, a_d, menc, zbuf, xh, gat, 64);
    finish12_k<<<cdiv(NNODE * 64, 256), 256, 0, stream>>>(gat, b1, lin, xb);

    // ---- layer 2: K=256, H*C=256, C=64 ----
    gemm(xb, Wb2, nullptr, xh, 256, 256);
    gemm(xb, Lb2, lb2,    lin, 256, 256);
    attn_k<<<cdiv(NNODE * HEADS, 256), 256, 0, stream>>>(xh, as2, ad2, a_s, a_d, 64);
    zero((float*)menc, NNODE * HEADS); zero(zbuf, NNODE * HEADS); zero(gat, NNODE * 256);
    edge_max_k<<<egrid, 256, 0, stream>>>(eidx, a_s, a_d, menc);
    edge_sum_k<<<egrid, 256, 0, stream>>>(eidx, a_s, a_d, menc, zbuf);
    edge_msg_k<<<cdiv(EPLUS, 16), 256, 0, stream>>>(eidx, a_s, a_d, menc, zbuf, xh, gat, 64);
    finish12_k<<<cdiv(NNODE * 64, 256), 256, 0, stream>>>(gat, b2, lin, xb);

    // ---- layer 3: K=256, H*C=512, C=128, mean over heads ----
    gemm(xb, Wb3, nullptr, xh, 256, 512);
    gemm(xb, Lb3, lb3,    lin, 256, 128);
    attn_k<<<cdiv(NNODE * HEADS, 256), 256, 0, stream>>>(xh, as3, ad3, a_s, a_d, 128);
    zero((float*)menc, NNODE * HEADS); zero(zbuf, NNODE * HEADS); zero(gat, NNODE * 512);
    edge_max_k<<<egrid, 256, 0, stream>>>(eidx, a_s, a_d, menc);
    edge_sum_k<<<egrid, 256, 0, stream>>>(eidx, a_s, a_d, menc, zbuf);
    edge_msg_k<<<cdiv(EPLUS, 8), 256, 0, stream>>>(eidx, a_s, a_d, menc, zbuf, xh, gat, 128);
    finish3_k<<<cdiv(NNODE * 128, 256), 256, 0, stream>>>(gat, b3, lin, xo);

    // ---- global mean pool ----
    zero(pooled, NGRAPH * 128); zero(cnt, NGRAPH);
    pool_acc_k<<<cdiv(NNODE * 128, 256), 256, 0, stream>>>(xo, batch, pooled, cnt);
    pool_div_k<<<cdiv(NGRAPH * 128, 256), 256, 0, stream>>>(pooled, cnt, out_side);
  };

  run_side(x_l, edge_l, batch_l, out);
  run_side(x_r, edge_r, batch_r, out + NGRAPH * 128);
}
